// MeshClassificationGNN_60043642798824
// MI455X (gfx1250) — matrix operand order repaired and checked by reading, not compiled, not run on hardware
//
#include <hip/hip_runtime.h>

typedef __bf16 bf16_t;
typedef __attribute__((ext_vector_type(16))) __bf16 v16bf;
typedef __attribute__((ext_vector_type(8)))  __bf16 v8bf;
typedef __attribute__((ext_vector_type(8)))  float  v8f;

#define DH   128
#define DOUT 64

// ---------------------------------------------------------------- degree
__global__ void gnn_deg_kernel(const int* __restrict__ dst, float* __restrict__ deg, int E) {
    int t = blockIdx.x * blockDim.x + threadIdx.x;
    if (t < E) atomicAdd(&deg[dst[t]], 1.0f);
}

__global__ void gnn_invdeg_kernel(float* __restrict__ deg, int N) {
    int t = blockIdx.x * blockDim.x + threadIdx.x;
    if (t < N) deg[t] = 1.0f / fmaxf(deg[t], 1.0f);
}

// ---------------------------------------------------------------- x -> bf16
__global__ void gnn_cast_x_kernel(const float* __restrict__ x, bf16_t* __restrict__ h, int n4) {
    int t = blockIdx.x * blockDim.x + threadIdx.x;
    if (t >= n4) return;
    float4 v = ((const float4*)x)[t];
    union { bf16_t b[4]; uint2 u; } tmp;
    tmp.b[0] = (bf16_t)v.x; tmp.b[1] = (bf16_t)v.y;
    tmp.b[2] = (bf16_t)v.z; tmp.b[3] = (bf16_t)v.w;
    ((uint2*)h)[t] = tmp.u;
}

// ---------------------------------------------------------------- weight pack
// Packed B layout per 32x16 K x N tile: lane l -> column n = l&15, K-half = l>>4,
// 16 contiguous bf16 = K (khalf*8 .. +7, then 16+khalf*8 .. +7) for that column. This matches
// the v_wmma_16x16x32 B operand VGPR striping, so the GEMM loads B with one
// 32-byte contiguous load per lane.
// Combined K=256: k-tiles 0..3 = Wl rows (agg branch), k-tiles 4..7 = Wr rows.
__global__ void gnn_pack_w_kernel(const float* __restrict__ Wl, const float* __restrict__ Wr,
                                  const float* __restrict__ Wll, const float* __restrict__ Wrl,
                                  bf16_t* __restrict__ Bpack) {
    int tid = blockIdx.x * blockDim.x + threadIdx.x;
    const int mainCnt = 8 * 8 * 8 * 32;            // layers x kt x nt x lane
    if (tid < mainCnt) {
        int lane = tid & 31;
        int nt   = (tid >> 5) & 7;
        int kt   = (tid >> 8) & 7;
        int l    = tid >> 11;
        int n     = nt * 16 + (lane & 15);
        int khalf = lane >> 4;
        bf16_t* dst = Bpack + ((size_t)(((l * 8 + kt) * 8 + nt) * 32 + lane) << 4);
        #pragma unroll
        for (int j = 0; j < 16; ++j) {
            int kg = kt * 32 + (j & 8) * 2 + khalf * 8 + (j & 7);     // 0..255 combined K
            float v = (kt < 4) ? Wl[((size_t)l * 128 + kg) * 128 + n]
                               : Wr[((size_t)l * 128 + (kg - 128)) * 128 + n];
            dst[j] = (bf16_t)v;
        }
    } else if (tid < mainCnt + 8 * 4 * 32) {       // last layer, N=64
        int t    = tid - mainCnt;
        int lane = t & 31;
        int nt   = (t >> 5) & 3;
        int kt   = t >> 7;                          // 0..7
        int n     = nt * 16 + (lane & 15);
        int khalf = lane >> 4;
        bf16_t* dst = Bpack + (size_t)262144 + ((size_t)((kt * 4 + nt) * 32 + lane) << 4);
        #pragma unroll
        for (int j = 0; j < 16; ++j) {
            int kg = kt * 32 + (j & 8) * 2 + khalf * 8 + (j & 7);
            float v = (kt < 4) ? Wll[(size_t)kg * 64 + n]
                               : Wrl[(size_t)(kg - 128) * 64 + n];
            dst[j] = (bf16_t)v;
        }
    }
}

// ---------------------------------------------------------------- scatter-add (mean numerator)
// One thread = one edge x 8 features: b128 bf16 gather (L2-resident) + 8 f32 atomics.
__global__ void gnn_scatter_kernel(const bf16_t* __restrict__ h, float* __restrict__ agg,
                                   const int* __restrict__ src, const int* __restrict__ dst, int E) {
    long t = (long)blockIdx.x * blockDim.x + threadIdx.x;
    long e = t >> 4;
    if (e >= E) return;
    int c = (int)(t & 15) << 3;
    int s = src[e], d = dst[e];
    v8bf v = *(const v8bf*)(h + (size_t)s * DH + c);
    float* ap = agg + (size_t)d * DH + c;
    #pragma unroll
    for (int j = 0; j < 8; ++j) atomicAdd(ap + j, (float)v[j]);
}

// ---------------------------------------------------------------- fused SAGE GEMM
// out[M,Nout] = act( [agg*invdeg | h] (M x 256) @ Bpack (256 x Nout) + bias )
// One wave handles a 16-row M tile across all NT n-tiles. K = 8 tiles of 32 (bf16 WMMA).
// NT/RELU/output type are compile-time so the accumulators stay in VGPRs (no
// dynamic indexing -> no scratch spills) and both loops fully unroll.
template <int NT, bool RELU, bool OUTBF16>
__global__ void gnn_sage_gemm_kernel(const float* __restrict__ agg,
                                     const float* __restrict__ invdeg,
                                     const bf16_t* __restrict__ h,
                                     const bf16_t* __restrict__ Bp,
                                     const float* __restrict__ bias,
                                     bf16_t* __restrict__ outB, float* __restrict__ outF,
                                     int mTiles) {
    int lane  = threadIdx.x & 31;
    int wave  = threadIdx.x >> 5;
    int mTile = blockIdx.x * 8 + wave;
    if (mTile >= mTiles) return;                    // wave-uniform: EXEC all-1s for WMMA
    int r    = lane & 15;                           // A-row / B-col / C-col within tile
    int half = lane >> 4;                           // K-half selector
    int row  = mTile * 16 + r;
    float s  = invdeg[row];

    v8f acc[NT];
    #pragma unroll
    for (int i = 0; i < NT; ++i)
        #pragma unroll
        for (int j = 0; j < 8; ++j) acc[i][j] = 0.0f;

    #pragma unroll
    for (int kt = 0; kt < 8; ++kt) {
        v16bf a;
        if (kt < 4) {                               // agg branch: f32 load, scale, cvt bf16
            const float* ap = agg + (size_t)row * DH + kt * 32 + half * 8;
            float4 f0 = *(const float4*)(ap);
            float4 f1 = *(const float4*)(ap + 4);
            float4 f2 = *(const float4*)(ap + 16);
            float4 f3 = *(const float4*)(ap + 20);
            a[0]  = (bf16_t)(f0.x * s); a[1]  = (bf16_t)(f0.y * s);
            a[2]  = (bf16_t)(f0.z * s); a[3]  = (bf16_t)(f0.w * s);
            a[4]  = (bf16_t)(f1.x * s); a[5]  = (bf16_t)(f1.y * s);
            a[6]  = (bf16_t)(f1.z * s); a[7]  = (bf16_t)(f1.w * s);
            a[8]  = (bf16_t)(f2.x * s); a[9]  = (bf16_t)(f2.y * s);
            a[10] = (bf16_t)(f2.z * s); a[11] = (bf16_t)(f2.w * s);
            a[12] = (bf16_t)(f3.x * s); a[13] = (bf16_t)(f3.y * s);
            a[14] = (bf16_t)(f3.z * s); a[15] = (bf16_t)(f3.w * s);
        } else {                                    // self branch: direct bf16 b128 loads
            const bf16_t* hp = h + (size_t)row * DH + (kt - 4) * 32 + half * 8;
            v8bf lo = *(const v8bf*)(hp);
            v8bf hi = *(const v8bf*)(hp + 16);
            #pragma unroll
            for (int j = 0; j < 8; ++j) { a[j] = lo[j]; a[8 + j] = hi[j]; }
        }
        #pragma unroll
        for (int nt = 0; nt < NT; ++nt) {
            v16bf bm = *(const v16bf*)(Bp + ((size_t)((kt * NT + nt) * 32 + lane) << 4));
            acc[nt] = __builtin_amdgcn_wmma_f32_16x16x32_bf16(
                false, a, false, bm, (short)0, acc[nt], false, false);
        }
    }

    const int ldo = NT * 16;
    #pragma unroll
    for (int nt = 0; nt < NT; ++nt) {
        int col  = nt * 16 + r;
        float bb = bias[col];
        #pragma unroll
        for (int v = 0; v < 8; ++v) {
            int orow  = mTile * 16 + half * 8 + v;  // C layout: VGPR v, lane-half selects M+8
            float val = acc[nt][v] + bb;
            if (RELU) val = fmaxf(val, 0.0f);
            if (OUTBF16) outB[(size_t)orow * ldo + col] = (bf16_t)val;
            else         outF[(size_t)orow * ldo + col] = val;
        }
    }
}

// ---------------------------------------------------------------- graph mean pool
__global__ void gnn_reduce_kernel(const float* __restrict__ hlast, float* __restrict__ out, int N) {
    int f   = threadIdx.x & 63;
    int sub = threadIdx.x >> 6;                     // 0..3
    float ssum = 0.0f;
    for (int row = blockIdx.x * 4 + sub; row < N; row += gridDim.x * 4)
        ssum += hlast[(size_t)row * DOUT + f];
    atomicAdd(&out[f], ssum * (1.0f / (float)N));
}

// ---------------------------------------------------------------- host
extern "C" void kernel_launch(void* const* d_in, const int* in_sizes, int n_in,
                              void* d_out, int out_size, void* d_ws, size_t ws_size,
                              hipStream_t stream) {
    const float* x   = (const float*)d_in[0];
    const int*   ei  = (const int*)d_in[1];
    const float* Wl  = (const float*)d_in[2];
    const float* Wr  = (const float*)d_in[3];
    const float* b   = (const float*)d_in[4];
    const float* Wll = (const float*)d_in[5];
    const float* Wrl = (const float*)d_in[6];
    const float* bl  = (const float*)d_in[7];

    int N = in_sizes[0] / DH;
    int E = in_sizes[1] / 2;
    const int* src  = ei;
    const int* dstp = ei + E;

    char*  ws  = (char*)d_ws;
    size_t off = 0;
    auto take = [&](size_t bytes) -> char* {
        char* p = ws + off;
        off = (off + bytes + 255) & ~(size_t)255;
        return p;
    };
    bf16_t* hA     = (bf16_t*)take((size_t)N * DH * 2);
    bf16_t* hB     = (bf16_t*)take((size_t)N * DH * 2);
    float*  agg    = (float*) take((size_t)N * DH * 4);
    float*  invdeg = (float*) take((size_t)N * 4);
    bf16_t* Bpack  = (bf16_t*)take((size_t)(8 * 32768 + 16384) * 2);
    float*  hlast  = (float*) take((size_t)N * DOUT * 4);
    (void)ws_size; (void)n_in; (void)out_size;

    // degree -> 1/max(deg,1)
    hipMemsetAsync(invdeg, 0, (size_t)N * 4, stream);
    gnn_deg_kernel<<<(E + 255) / 256, 256, 0, stream>>>(dstp, invdeg, E);
    gnn_invdeg_kernel<<<(N + 255) / 256, 256, 0, stream>>>(invdeg, N);

    // x -> bf16 feature matrix; pack all weights into WMMA-B layout
    int n4 = N * DH / 4;
    gnn_cast_x_kernel<<<(n4 + 255) / 256, 256, 0, stream>>>(x, hA, n4);
    gnn_pack_w_kernel<<<(17408 + 255) / 256, 256, 0, stream>>>(Wl, Wr, Wll, Wrl, Bpack);

    int mTiles = (N + 15) / 16;
    int gemmBlocks = (mTiles + 7) / 8;
    bf16_t* cur = hA; bf16_t* nxt = hB;

    for (int l = 0; l < 9; ++l) {
        hipMemsetAsync(agg, 0, (size_t)N * DH * 4, stream);
        long st = (long)E * 16;
        gnn_scatter_kernel<<<(unsigned)((st + 255) / 256), 256, 0, stream>>>(cur, agg, src, dstp, E);
        if (l < 8) {
            gnn_sage_gemm_kernel<8, true, true><<<gemmBlocks, 256, 0, stream>>>(
                agg, invdeg, cur, Bpack + (size_t)l * 32768, b + (size_t)l * DH,
                nxt, nullptr, mTiles);
            bf16_t* t = cur; cur = nxt; nxt = t;
        } else {
            gnn_sage_gemm_kernel<4, false, false><<<gemmBlocks, 256, 0, stream>>>(
                agg, invdeg, cur, Bpack + (size_t)262144, bl,
                nullptr, hlast, mTiles);
        }
    }

    hipMemsetAsync(d_out, 0, 64 * sizeof(float), stream);
    gnn_reduce_kernel<<<256, 256, 0, stream>>>(hlast, (float*)d_out, N);
}
